// VQVAE_62371515073002
// MI455X (gfx1250) — compile-verified, matrix operand order checked by reading, not run on hardware
//
#include <hip/hip_runtime.h>
#include <hip/hip_bf16.h>

typedef __attribute__((ext_vector_type(16))) _Float16 v16h;
typedef __attribute__((ext_vector_type(8)))  _Float16 v8h;
typedef __attribute__((ext_vector_type(8)))  float    v8f;

// Problem constants: x [B=8, E=64, D=32, H=32, W=32], embedding [64, 512]
// N = B*W*D*H = 262144, flat = [N, 64], K = 512.
#define VQ_N      262144
#define VQ_E      64
#define VQ_K      512
#define VQ_DECAY  0.99f
#define VQ_EPS    1e-5f

// GEMM staging: embT row padded 64 -> 72 halves (36-dword lane stride, 16
// lanes hit 16 distinct banks on ds_load_b128).
#define SB_STRIDE      72
#define SB_BYTES       (VQ_K * SB_STRIDE * 2)            // 73728
#define GEMM_LDS_BYTES (SB_BYTES + VQ_K * 4)             // + e2 = 75776
#define TILES_PER_WAVE 4

// Gather partial-sum config
#define GATHER_BLOCKS   128
#define GROUPS_PER_BLK  64            // 8192 (b,d,h) groups / 128 blocks
#define PART_STRIDE     (VQ_E * VQ_K + VQ_K)             // 33280 floats
#define GATHER_LDS_BYTES (VQ_E * VQ_K * 4 + VQ_K * 4 + 256 * 4) // 134144

// ---------------------------------------------------------------------------
// 1) codebook prep: embT[k][e] = (f16) embedding[e][k]  (one WMMA-B column is
//    a contiguous 64-f16 row), plus e2[k] = sum_e embedding[e][k]^2 (f32).
// ---------------------------------------------------------------------------
__global__ void vq_prep_e(const float* __restrict__ emb,
                          _Float16* __restrict__ embT,
                          float* __restrict__ e2) {
    int k = blockIdx.x * blockDim.x + threadIdx.x;
    if (k >= VQ_K) return;
    float acc = 0.0f;
    for (int e = 0; e < VQ_E; ++e) {
        float v = emb[e * VQ_K + k];
        embT[(size_t)k * VQ_E + e] = (_Float16)v;
        acc += v * v;
    }
    e2[k] = acc;
}

// ---------------------------------------------------------------------------
// 2) x transpose+convert: xf16[n][e], n = ((b*32+w)*32+d)*32+h.
//    LDS-tiled: coalesced 128B loads over w, bank-padded on the e dim.
//    Grid: 8 e-chunks * 32 d * 8 b = 2048 blocks of 256.
// ---------------------------------------------------------------------------
__global__ __launch_bounds__(256) void vq_prep_x(const float* __restrict__ x,
                                                 _Float16* __restrict__ xf16) {
    __shared__ float tile[32][32][9];     // [h][w][e_local], 9 pads banks
    const int bid = blockIdx.x;
    const int ec  = bid & 7;              // e-chunk of 8
    const int d   = (bid >> 3) & 31;
    const int b   = bid >> 8;
    const int tid = threadIdx.x;
    for (int j = 0; j < 32; ++j) {        // 8192 f32, consecutive tid -> w
        int lin = tid + j * 256;
        int e_l = lin >> 10;
        int rem = lin & 1023;
        int h = rem >> 5;
        int w = rem & 31;
        int e = (ec << 3) + e_l;
        tile[h][w][e_l] =
            x[(size_t)b * 2097152 + (size_t)e * 32768 + d * 1024 + h * 32 + w];
    }
    __syncthreads();
    for (int j = 0; j < 4; ++j) {         // 8 contiguous f16 = 16B per store
        int lin = tid + j * 256;
        int w = lin >> 5;
        int h = lin & 31;
        int n = b * 32768 + w * 1024 + d * 32 + h;
        _Float16* dst = xf16 + (size_t)n * VQ_E + (ec << 3);
        v8h o;
        #pragma unroll
        for (int e = 0; e < 8; ++e) o[e] = (_Float16)tile[h][w][e];
        *(v8h*)dst = o;
    }
}

// ---------------------------------------------------------------------------
// 3) distance GEMM + argmin. embT + e2 staged once per block in dynamic LDS
//    (shared by 8 waves); each wave handles 4 row-tiles of 16 rows x all 512
//    codes. Per 16-col tile: 2x v_wmma_f32_16x16x32_f16 (K = 64 = 2*32),
//    dist = e2[col] - 2*acc, running argmin in the native C layout
//    (vgpr r <-> row r + 8*(lane>=16), N = lane&15), then a 16-lane xor-shuffle
//    argmin reduction (masks 1..8 never cross the 16-lane groups, so
//    full-width shfl_xor is equivalent and lowers to plain swizzles).
//    ISA 16-bit A/B layout: lane L(<16): K{0..7,16..23}; lane L+16: K{8..15,24..31}.
// ---------------------------------------------------------------------------
__global__ __launch_bounds__(256) void vq_gemm_argmin(
        const _Float16* __restrict__ xf16, const _Float16* __restrict__ embT,
        const float* __restrict__ e2, int* __restrict__ idx) {
    extern __shared__ unsigned char smem[];
    _Float16* sB  = (_Float16*)smem;                 // 512 * 72 halves
    float*    sE2 = (float*)(smem + SB_BYTES);       // 512 f32
    const int tid  = threadIdx.x;
    const int lane = tid & 31;
    const int wv   = tid >> 5;                       // wave in block, 0..7
    const int lr   = lane & 15;
    const int hi   = lane >> 4;

    #pragma unroll
    for (int j = 0; j < 16; ++j) {                   // 4096 chunks of 8 halves
        int i = tid + j * 256;
        int row = i >> 3, c = i & 7;
        *(v8h*)(sB + row * SB_STRIDE + c * 8) =
            *(const v8h*)(embT + row * VQ_E + c * 8);
    }
    sE2[tid]       = e2[tid];
    sE2[tid + 256] = e2[tid + 256];
    __syncthreads();

    union V16 { v16h v; v8h h[2]; };

    for (int t = 0; t < TILES_PER_WAVE; ++t) {
        const int nbase = ((blockIdx.x * 8 + wv) * TILES_PER_WAVE + t) * 16;
        const _Float16* ap = xf16 + (size_t)(nbase + lr) * VQ_E + (hi << 3);
        V16 a0, a1;
        a0.h[0] = *(const v8h*)(ap + 0);             // K 0..7   (or 8..15)
        a0.h[1] = *(const v8h*)(ap + 16);            // K 16..23 (or 24..31)
        a1.h[0] = *(const v8h*)(ap + 32);            // chunk E=32..63
        a1.h[1] = *(const v8h*)(ap + 48);

        float mind[8];
        int   mini[8];
        #pragma unroll
        for (int r = 0; r < 8; ++r) { mind[r] = 3.402823466e38f; mini[r] = 0; }

        #pragma unroll 2
        for (int ct = 0; ct < VQ_K / 16; ++ct) {
            const int col = (ct << 4) + lr;
            const _Float16* bp = sB + col * SB_STRIDE + (hi << 3);
            V16 b0, b1;
            b0.h[0] = *(const v8h*)(bp + 0);
            b0.h[1] = *(const v8h*)(bp + 16);
            b1.h[0] = *(const v8h*)(bp + 32);
            b1.h[1] = *(const v8h*)(bp + 48);

            v8f c = {0.f, 0.f, 0.f, 0.f, 0.f, 0.f, 0.f, 0.f};
            c = __builtin_amdgcn_wmma_f32_16x16x32_f16(false, a0.v, false, b0.v,
                                                       (short)0, c, false, false);
            c = __builtin_amdgcn_wmma_f32_16x16x32_f16(false, a1.v, false, b1.v,
                                                       (short)0, c, false, false);
            const float e2c = sE2[col];
            #pragma unroll
            for (int r = 0; r < 8; ++r) {
                float dist = fmaf(-2.0f, c[r], e2c);
                if (dist < mind[r]) { mind[r] = dist; mini[r] = col; }
            }
        }
        #pragma unroll
        for (int off = 8; off >= 1; off >>= 1) {
            #pragma unroll
            for (int r = 0; r < 8; ++r) {
                float od = __shfl_xor(mind[r], off);
                int   oi = __shfl_xor(mini[r], off);
                if (od < mind[r] || (od == mind[r] && oi < mini[r])) {
                    mind[r] = od; mini[r] = oi;
                }
            }
        }
        if (lr == 0) {
            #pragma unroll
            for (int r = 0; r < 8; ++r) idx[nbase + hi * 8 + r] = mini[r];
        }
    }
}

// ---------------------------------------------------------------------------
// 4) gather + outputs + segment sums. 128 blocks, each streams 64 (b,d,h)
//    slabs with w-contiguous (coalesced) x/out access; emb_sum/counts are
//    accumulated in LDS (ds_add atomics, no global atomics), then flushed as
//    per-block partials with plain stores.
// ---------------------------------------------------------------------------
__global__ __launch_bounds__(256) void vq_gather(
        const float* __restrict__ x, const float* __restrict__ emb,
        const int* __restrict__ idx, float* __restrict__ out,
        float* __restrict__ code_out, float* __restrict__ part,
        float* __restrict__ diffp) {
    extern __shared__ unsigned char smem[];
    float*        s_emb = (float*)smem;                           // 64*512 f32
    unsigned int* s_cnt = (unsigned int*)(smem + VQ_E * VQ_K * 4);// 512 u32
    float*        s_red = (float*)(smem + VQ_E * VQ_K * 4 + VQ_K * 4);

    const int tid = threadIdx.x;
    for (int i = tid; i < VQ_E * VQ_K; i += 256) s_emb[i] = 0.0f;
    for (int i = tid; i < VQ_K; i += 256)        s_cnt[i] = 0u;
    __syncthreads();

    const int w  = tid & 31;
    const int eg = tid >> 5;                      // 8 e-groups, 8 e's each
    float sq = 0.0f;

    const int g0 = blockIdx.x * GROUPS_PER_BLK;
    for (int g = g0; g < g0 + GROUPS_PER_BLK; ++g) {
        const int h = g & 31;
        const int d = (g >> 5) & 31;
        const int b = g >> 10;
        const int n = b * 32768 + w * 1024 + d * 32 + h;
        const int k = idx[n];
        const size_t base = (size_t)b * 2097152 + (size_t)d * 1024 + h * 32 + w;
        #pragma unroll
        for (int j = 0; j < 8; ++j) {
            int e = eg + j * 8;
            float q  = emb[e * VQ_K + k];
            size_t o = base + (size_t)e * 32768;
            float xv = x[o];
            out[o] = q;                           // straight-through == quantize
            float dd = q - xv;
            sq += dd * dd;
            atomicAdd(&s_emb[e * VQ_K + k], xv);  // ds_add_f32
        }
        if (eg == 0) {
            code_out[n] = (float)k;
            atomicAdd(&s_cnt[k], 1u);
        }
    }
    __syncthreads();

    float* pb = part + (size_t)blockIdx.x * PART_STRIDE;
    for (int i = tid; i < VQ_E * VQ_K; i += 256) pb[i] = s_emb[i];
    for (int i = tid; i < VQ_K; i += 256) pb[VQ_E * VQ_K + i] = (float)s_cnt[i];

    s_red[tid] = sq;
    __syncthreads();
    for (int s = 128; s > 0; s >>= 1) {
        if (tid < s) s_red[tid] += s_red[tid + s];
        __syncthreads();
    }
    if (tid == 0) diffp[blockIdx.x] = s_red[0];
}

// ---------------------------------------------------------------------------
// 5) reduce the 128 per-block partials -> emb_sumF [64*512], countsF [512]
// ---------------------------------------------------------------------------
__global__ __launch_bounds__(256) void vq_reduce(const float* __restrict__ part,
                                                 float* __restrict__ emb_sumF,
                                                 float* __restrict__ countsF) {
    int i = blockIdx.x * 256 + threadIdx.x;
    if (i >= PART_STRIDE) return;
    float s = 0.0f;
    for (int b = 0; b < GATHER_BLOCKS; ++b)
        s += part[(size_t)b * PART_STRIDE + i];
    if (i < VQ_E * VQ_K) emb_sumF[i] = s;
    else                 countsF[i - VQ_E * VQ_K] = s;
}

// ---------------------------------------------------------------------------
// 6) EMA + normalize epilogue. One block, 512 threads (k = tid).
// ---------------------------------------------------------------------------
__global__ __launch_bounds__(512) void vq_finalize(
        const float* __restrict__ cluster_size, const float* __restrict__ emb_mean,
        const float* __restrict__ emb_sumF, const float* __restrict__ countsF,
        const float* __restrict__ diffp,
        float* __restrict__ d_ncs, float* __restrict__ d_nem,
        float* __restrict__ d_nemb, float* __restrict__ d_diff) {
    const int k = threadIdx.x;
    float cnt = countsF[k];
    float ncs = cluster_size[k] * VQ_DECAY + (1.0f - VQ_DECAY) * cnt;
    d_ncs[k] = ncs;
    __shared__ float red[512];
    red[k] = ncs;
    __syncthreads();
    for (int s = 256; s > 0; s >>= 1) {
        if (k < s) red[k] += red[k + s];
        __syncthreads();
    }
    float nsum = red[0];
    __syncthreads();
    // reduce diff partials
    red[k] = (k < GATHER_BLOCKS) ? diffp[k] : 0.0f;
    __syncthreads();
    for (int s = 256; s > 0; s >>= 1) {
        if (k < s) red[k] += red[k + s];
        __syncthreads();
    }
    float cs = (ncs + VQ_EPS) / (nsum + (float)VQ_K * VQ_EPS) * nsum;
    float inv = 1.0f / cs;
    for (int e = 0; e < VQ_E; ++e) {
        float em = emb_mean[e * VQ_K + k] * VQ_DECAY +
                   (1.0f - VQ_DECAY) * emb_sumF[e * VQ_K + k];
        d_nem[e * VQ_K + k]  = em;
        d_nemb[e * VQ_K + k] = em * inv;
    }
    if (k == 0) d_diff[0] = 0.25f * red[0] / (float)(VQ_N * VQ_E);
}

// ---------------------------------------------------------------------------
extern "C" void kernel_launch(void* const* d_in, const int* in_sizes, int n_in,
                              void* d_out, int out_size, void* d_ws, size_t ws_size,
                              hipStream_t stream) {
    (void)in_sizes; (void)n_in; (void)out_size; (void)ws_size;
    const float* x     = (const float*)d_in[0];   // [8,64,32,32,32]
    const float* emb   = (const float*)d_in[1];   // [64,512]
    const float* csz   = (const float*)d_in[2];   // [512]
    const float* emean = (const float*)d_in[3];   // [64,512]

    float* out    = (float*)d_out;                // [8,64,32,32,32] = 16777216
    float* o_diff = out + 16777216;               // scalar
    float* o_code = out + 16777217;               // [N] = 262144 (as float)
    float* o_nemb = out + 17039361;               // new_embedding [64,512]
    float* o_ncs  = out + 17072129;               // new_cluster_size [512]
    float* o_nem  = out + 17072641;               // new_embedding_mean [64,512]

    unsigned char* w = (unsigned char*)d_ws;
    _Float16* xf16 = (_Float16*)w;   w += (size_t)VQ_N * VQ_E * 2;          // 32 MB
    _Float16* embT = (_Float16*)w;   w += (size_t)VQ_K * VQ_E * 2;          // 64 KB
    float* e2      = (float*)w;      w += (size_t)VQ_K * 4;
    int* idx       = (int*)w;        w += (size_t)VQ_N * 4;                 // 1 MB
    float* part    = (float*)w;      w += (size_t)GATHER_BLOCKS * PART_STRIDE * 4; // 17 MB
    float* diffp   = (float*)w;      w += (size_t)GATHER_BLOCKS * 4;
    float* emb_sumF= (float*)w;      w += (size_t)VQ_E * VQ_K * 4;          // 128 KB
    float* countsF = (float*)w;      w += (size_t)VQ_K * 4;

    hipLaunchKernelGGL(vq_prep_e, dim3(2), dim3(256), 0, stream, emb, embT, e2);
    hipLaunchKernelGGL(vq_prep_x, dim3(2048), dim3(256), 0, stream, x, xf16);
    hipLaunchKernelGGL(vq_gemm_argmin, dim3(512), dim3(256), GEMM_LDS_BYTES,
                       stream, xf16, embT, e2, idx);
    hipLaunchKernelGGL(vq_gather, dim3(GATHER_BLOCKS), dim3(256),
                       GATHER_LDS_BYTES, stream, x, emb, idx, out, o_code,
                       part, diffp);
    hipLaunchKernelGGL(vq_reduce, dim3((PART_STRIDE + 255) / 256), dim3(256),
                       0, stream, part, emb_sumF, countsF);
    hipLaunchKernelGGL(vq_finalize, dim3(1), dim3(512), 0, stream,
                       csz, emean, emb_sumF, countsF, diffp,
                       o_ncs, o_nem, o_nemb, o_diff);
}
